// SAGPool_56573309223700
// MI455X (gfx1250) — compile-verified, matrix operand order checked by reading, not run on hardware
//
#include <hip/hip_runtime.h>
#include <hip/hip_bf16.h>

typedef __attribute__((ext_vector_type(16))) _Float16 v16h;
typedef __attribute__((ext_vector_type(8)))  float    v8f;

// ---- fixed problem geometry -------------------------------------------------
#define NUM_GRAPHS       128
#define NODES_PER_GRAPH  1024
#define NUM_NODES        (NUM_GRAPHS * NODES_PER_GRAPH)   // 131072
#define NUM_EDGES        (NUM_NODES * 32)                 // 4194304
#define CHANNELS         256
#define KTOP             512
#define NPOOL            (NUM_GRAPHS * KTOP)              // 65536

// ---- output layout (flat f32 concat, reference return order) ---------------
static constexpr size_t X_OUT_OFF  = 0;                                   // 65536*256
static constexpr size_t EDGE_OFF   = (size_t)NPOOL * CHANNELS;            // 16777216
static constexpr size_t BATCH_OFF  = EDGE_OFF + (size_t)2 * NUM_EDGES;    // 25165824
static constexpr size_t PERM_OFF   = BATCH_OFF + NPOOL;                   // 25231360
static constexpr size_t SCOREP_OFF = PERM_OFF + NPOOL;                    // 25296896

// Branchless tanh: 1 - 2/(exp2(2x*log2e)+1). Monotone, saturates correctly at
// +/-inf, no exec divergence (v_exp_f32 + v_rcp_f32), TRANS co-executes with XDL.
__device__ __forceinline__ float fast_tanh(float x) {
  float e = __builtin_amdgcn_exp2f(x * 2.8853900817779268f);   // e^(2x)
  return 1.0f - 2.0f * __builtin_amdgcn_rcpf(e + 1.0f);
}

// =============================================================================
// Kernel 0: convert W1 (f32, [K=256][N=256] row-major) into f16, pre-swizzled
// into WMMA B-fragment order so each lane reads one contiguous v16h.
// B layout (16-bit, 32x16): lane n=lane%16, hi=lane>=16;
//   h[j] (j=0..15): chunk=j>>3, k = kstep*32 + chunk*16 + (hi?8:0) + (j&7)
// =============================================================================
__global__ void w1_to_f16_swizzled(const float* __restrict__ W1,
                                   _Float16* __restrict__ w1h) {
  int t    = blockIdx.x * 256 + threadIdx.x;   // 0..65535
  int j    = t & 15;
  int lane = (t >> 4) & 31;
  int ks   = (t >> 9) & 7;
  int nb   = t >> 12;
  int hi   = (lane >= 16) ? 8 : 0;
  int n    = nb * 16 + (lane & 15);
  int k    = ks * 32 + ((j >> 3) << 4) + hi + (j & 7);
  w1h[t]   = (_Float16)W1[k * CHANNELS + n];
}

// =============================================================================
// Kernel 1: scores = tanh(x@W1 + b1) @ W2 + b2 via v_wmma_f32_16x16x32_f16.
// 512 blocks x 512 threads (16 waves). One wave owns one 16-row tile of x,
// keeps the full 16x256 A slice in registers (8 v16h frags, x read ONCE), and
// loops the 16 N-blocks; W1(f16) staged through LDS in two 64KB phases.
// B fragments are software-pipelined 2 deep so ds_load latency overlaps WMMA.
// =============================================================================
__global__ __launch_bounds__(512) void score_gemm(
    const float* __restrict__ x, const _Float16* __restrict__ w1h,
    const float* __restrict__ b1, const float* __restrict__ w2,
    const float* __restrict__ b2, float* __restrict__ scores)
{
  __shared__ _Float16 w1s[CHANNELS * 128];   // 32768 halves = 64 KB (half of W1)

  const int tid     = threadIdx.x;
  const int wave    = tid >> 5;
  const int lane    = tid & 31;
  const int hi      = (lane >= 16) ? 1 : 0;
  const int rowTile = blockIdx.x * 16 + wave;        // 0..8191
  const int m       = rowTile * 16 + (lane & 15);

  // A fragments: 16-bit A 16x32 layout => per lane two 8-wide K chunks.
  v16h A[8];
  const float4* xr4 = (const float4*)(x + (size_t)m * CHANNELS);
  #pragma unroll
  for (int ks = 0; ks < 8; ++ks) {
    const int c0 = ks * 32 + hi * 8;                 // multiple of 8
    float4 f0 = xr4[(c0 >> 2) + 0];
    float4 f1 = xr4[(c0 >> 2) + 1];
    float4 f2 = xr4[(c0 >> 2) + 4];                  // chunk1 = c0 + 16
    float4 f3 = xr4[(c0 >> 2) + 5];
    v16h a;
    a[0]=(_Float16)f0.x;  a[1]=(_Float16)f0.y;  a[2]=(_Float16)f0.z;  a[3]=(_Float16)f0.w;
    a[4]=(_Float16)f1.x;  a[5]=(_Float16)f1.y;  a[6]=(_Float16)f1.z;  a[7]=(_Float16)f1.w;
    a[8]=(_Float16)f2.x;  a[9]=(_Float16)f2.y;  a[10]=(_Float16)f2.z; a[11]=(_Float16)f2.w;
    a[12]=(_Float16)f3.x; a[13]=(_Float16)f3.y; a[14]=(_Float16)f3.z; a[15]=(_Float16)f3.w;
    A[ks] = a;
  }

  float sums[8];
  #pragma unroll
  for (int v = 0; v < 8; ++v) sums[v] = 0.f;

  for (int phase = 0; phase < 2; ++phase) {
    // Cooperative stage: 32768 halves = 4096 uint4; 512 threads x 8.
    const uint4* src = (const uint4*)(w1h + phase * (CHANNELS * 128));
    uint4* dst = (uint4*)w1s;
    #pragma unroll
    for (int i = 0; i < 8; ++i) dst[tid + i * 512] = src[tid + i * 512];
    __syncthreads();

    const v16h* wfrag = (const v16h*)w1s;
    for (int nbl = 0; nbl < 8; ++nbl) {
      const int   nb  = phase * 8 + nbl;
      const int   nn  = nb * 16 + (lane & 15);
      const float w2n = w2[nn];
      const float b1n = b1[nn];

      // 2-deep pipelined B-fragment stream from LDS.
      v8f  acc  = {};
      v16h bcur = wfrag[(nbl * 8 + 0) * 32 + lane];
      #pragma unroll
      for (int ks = 0; ks < 8; ++ks) {
        v16h bnext = (ks < 7) ? wfrag[(nbl * 8 + ks + 1) * 32 + lane] : bcur;
        acc = __builtin_amdgcn_wmma_f32_16x16x32_f16(
            /*neg_a=*/false, A[ks], /*neg_b=*/false, bcur,
            /*c_mod=*/(short)0, acc, /*reuse_a=*/false, /*reuse_b=*/false);
        bcur = bnext;
      }
      // C/D layout: VGPR v = row v (lanes 0-15) / row v+8 (lanes 16-31), col nn.
      #pragma unroll
      for (int v = 0; v < 8; ++v) {
        float h = fast_tanh(acc[v] + b1n);
        sums[v] += h * w2n;
      }
    }
    __syncthreads();
  }

  // Reduce each row-sum across its 16-lane half.
  #pragma unroll
  for (int off = 1; off < 16; off <<= 1) {
    #pragma unroll
    for (int v = 0; v < 8; ++v) sums[v] += __shfl_xor(sums[v], off, 32);
  }
  if ((lane & 15) == 0) {
    const float bias  = b2[0];
    const int   rbase = rowTile * 16 + hi * 8;
    #pragma unroll
    for (int v = 0; v < 8; ++v) scores[rbase + v] = sums[v] + bias;
  }
}

// =============================================================================
// Kernel 2: per-graph top-512 via LDS bitonic sort (descending, ties -> lower
// index first, matching lax.top_k). Writes perm/scores/batch outputs + keep.
// =============================================================================
__global__ __launch_bounds__(1024) void topk_pool(
    const float* __restrict__ scores, int* __restrict__ perm_i,
    int* __restrict__ keep, float* __restrict__ out)
{
  __shared__ float s[NODES_PER_GRAPH];
  __shared__ int   si[NODES_PER_GRAPH];
  const int g = blockIdx.x, t = threadIdx.x;
  s[t]  = scores[g * NODES_PER_GRAPH + t];
  si[t] = t;
  keep[g * NODES_PER_GRAPH + t] = 0;
  __syncthreads();

  for (int k = 2; k <= NODES_PER_GRAPH; k <<= 1) {
    for (int j = k >> 1; j > 0; j >>= 1) {
      int ixj = t ^ j;
      if (ixj > t) {
        float sa = s[t], sb = s[ixj];
        int   ia = si[t], ib = si[ixj];
        bool dsc    = ((t & k) == 0);                         // descending run
        bool aPrecB = (sa > sb) || (sa == sb && ia < ib);
        bool dosw   = dsc ? !aPrecB : aPrecB;
        if (dosw) { s[t] = sb; si[t] = ib; s[ixj] = sa; si[ixj] = ia; }
      }
      __syncthreads();
    }
  }

  if (t < KTOP) {
    int idx  = si[t];
    int gidx = g * NODES_PER_GRAPH + idx;
    int p    = g * KTOP + t;
    perm_i[p]            = gidx;
    keep[gidx]           = 1;
    out[PERM_OFF + p]    = (float)gidx;   // < 2^24: exact in f32
    out[SCOREP_OFF + p]  = s[t];
    out[BATCH_OFF + p]   = (float)g;
  }
}

// =============================================================================
// Kernel 3: x_pooled = x[perm]  (pure bandwidth, float4)
// =============================================================================
__global__ void gather_x(const float* __restrict__ x,
                         const int* __restrict__ perm_i,
                         float* __restrict__ out) {
  int row = blockIdx.x;
  int src = perm_i[row];
  const float4* a = (const float4*)(x + (size_t)src * CHANNELS);
  float4* o = (float4*)(out + X_OUT_OFF + (size_t)row * CHANNELS);
  o[threadIdx.x] = a[threadIdx.x];
}

// =============================================================================
// Kernel 4a: per-block (1024 edges) kept-edge counts
// =============================================================================
__global__ __launch_bounds__(256) void edge_count(
    const int* __restrict__ ei, const int* __restrict__ keep,
    int* __restrict__ bcnt)
{
  __shared__ int red[256];
  const int tid  = threadIdx.x;
  const int base = blockIdx.x * 1024 + tid * 4;
  int cnt = 0;
  #pragma unroll
  for (int i = 0; i < 4; ++i) {
    int e = base + i;
    int r = ei[e], c = ei[NUM_EDGES + e];
    cnt += (keep[r] & keep[c]);
  }
  red[tid] = cnt; __syncthreads();
  for (int s = 128; s > 0; s >>= 1) {
    if (tid < s) red[tid] += red[tid + s];
    __syncthreads();
  }
  if (tid == 0) bcnt[blockIdx.x] = red[0];
}

// =============================================================================
// Kernel 4b: single-block exclusive scan of the 4096 block counts
// =============================================================================
__global__ __launch_bounds__(1024) void scan_blocks(
    const int* __restrict__ bcnt, int* __restrict__ boff)
{
  __shared__ int sc[1024];
  const int t = threadIdx.x;
  int c0 = bcnt[t*4+0], c1 = bcnt[t*4+1], c2 = bcnt[t*4+2], c3 = bcnt[t*4+3];
  int tot = c0 + c1 + c2 + c3;
  sc[t] = tot; __syncthreads();
  for (int off = 1; off < 1024; off <<= 1) {
    int v   = sc[t];
    int add = (t >= off) ? sc[t - off] : 0;
    __syncthreads();
    sc[t] = v + add;
    __syncthreads();
  }
  int excl = sc[t] - tot;
  boff[t*4+0] = excl;
  boff[t*4+1] = excl + c0;
  boff[t*4+2] = excl + c0 + c1;
  boff[t*4+3] = excl + c0 + c1 + c2;
}

// =============================================================================
// Kernel 4c: fill edge_index_pooled region with -1
// =============================================================================
__global__ void fill_edges(float* __restrict__ out) {
  size_t i = (size_t)blockIdx.x * 256 + threadIdx.x;
  float4* o = (float4*)(out + EDGE_OFF);
  o[i] = make_float4(-1.f, -1.f, -1.f, -1.f);
}

// =============================================================================
// Kernel 4d: stable scatter of kept edges to the front (block-ordered offsets
// + in-block scan preserve original edge order)
// =============================================================================
__global__ __launch_bounds__(256) void edge_scatter(
    const int* __restrict__ ei, const int* __restrict__ keep,
    const int* __restrict__ boff, float* __restrict__ out)
{
  __shared__ int sc[256];
  const int tid  = threadIdx.x;
  const int base = blockIdx.x * 1024 + tid * 4;
  int r[4], c[4], f[4], cnt = 0;
  #pragma unroll
  for (int i = 0; i < 4; ++i) {
    int e = base + i;
    r[i] = ei[e];
    c[i] = ei[NUM_EDGES + e];
    f[i] = keep[r[i]] & keep[c[i]];
    cnt += f[i];
  }
  sc[tid] = cnt; __syncthreads();
  for (int off = 1; off < 256; off <<= 1) {
    int v   = sc[tid];
    int add = (tid >= off) ? sc[tid - off] : 0;
    __syncthreads();
    sc[tid] = v + add;
    __syncthreads();
  }
  int pos = boff[blockIdx.x] + sc[tid] - cnt;
  #pragma unroll
  for (int i = 0; i < 4; ++i) {
    if (f[i]) {
      out[EDGE_OFF + pos]             = (float)r[i];
      out[EDGE_OFF + NUM_EDGES + pos] = (float)c[i];
      ++pos;
    }
  }
}

// =============================================================================
extern "C" void kernel_launch(void* const* d_in, const int* in_sizes, int n_in,
                              void* d_out, int out_size, void* d_ws, size_t ws_size,
                              hipStream_t stream) {
  const float* x  = (const float*)d_in[0];
  const int*   ei = (const int*)d_in[1];        // [2][NUM_EDGES]
  // d_in[2] = batch (unused; batch_pooled is derived analytically)
  const float* W1 = (const float*)d_in[3];
  const float* b1 = (const float*)d_in[4];
  const float* W2 = (const float*)d_in[5];
  const float* b2 = (const float*)d_in[6];
  float* out = (float*)d_out;

  char* ws = (char*)d_ws;
  float*    scores = (float*)(ws + 0);           // 131072 f32  (512 KB)
  _Float16* w1h    = (_Float16*)(ws + 524288);   // 65536 f16   (128 KB)
  int*      perm_i = (int*)(ws + 655360);        // 65536 i32   (256 KB)
  int*      keep   = (int*)(ws + 917504);        // 131072 i32  (512 KB)
  int*      bcnt   = (int*)(ws + 1441792);       // 4096 i32
  int*      boff   = (int*)(ws + 1458176);       // 4096 i32

  hipLaunchKernelGGL(w1_to_f16_swizzled, dim3(256),   dim3(256),  0, stream, W1, w1h);
  hipLaunchKernelGGL(score_gemm,         dim3(512),   dim3(512),  0, stream, x, w1h, b1, W2, b2, scores);
  hipLaunchKernelGGL(topk_pool,          dim3(128),   dim3(1024), 0, stream, scores, perm_i, keep, out);
  hipLaunchKernelGGL(gather_x,           dim3(65536), dim3(64),   0, stream, x, perm_i, out);
  hipLaunchKernelGGL(edge_count,         dim3(4096),  dim3(256),  0, stream, ei, keep, bcnt);
  hipLaunchKernelGGL(scan_blocks,        dim3(1),     dim3(1024), 0, stream, bcnt, boff);
  hipLaunchKernelGGL(fill_edges,         dim3(8192),  dim3(256),  0, stream, out);
  hipLaunchKernelGGL(edge_scatter,       dim3(4096),  dim3(256),  0, stream, ei, keep, boff, out);
}
